// RenovateNet_Fine_10694468567627
// MI455X (gfx1250) — matrix operand order repaired and checked by reading, not compiled, not run on hardware
//
#include <hip/hip_runtime.h>
#include <hip/hip_bf16.h>
#include <math.h>

// ---------------- CDNA5 WMMA types ----------------
typedef __attribute__((ext_vector_type(16))) __bf16 v16bf;
typedef __attribute__((ext_vector_type(8)))  __bf16 v8bf;
typedef __attribute__((ext_vector_type(4)))  __bf16 v4bf;
typedef __attribute__((ext_vector_type(8)))  float  v8f;

#define N_CLASS 52
#define KPAD    64          // class dim padded to 4x16 WMMA tiles
#define N_COARSE 7
#define HDIM    256         // H == C == 256
#define ALP     0.125f
#define INV_TMP 8.0f        // 1/0.125
#define MOM     0.9f

// combine two 8-element bf16 LDS loads into one 16-wide WMMA fragment
__device__ __forceinline__ v16bf frag16(v8bf lo, v8bf hi)
{
    return __builtin_shufflevector(lo, hi, 0, 1, 2, 3, 4, 5, 6, 7,
                                   8, 9, 10, 11, 12, 13, 14, 15);
}

__device__ __forceinline__ v4bf cvt4(float4 v, float s)
{
    v4bf r;
    r[0] = (__bf16)(v.x * s); r[1] = (__bf16)(v.y * s);
    r[2] = (__bf16)(v.z * s); r[3] = (__bf16)(v.w * s);
    return r;
}

// =====================================================================
// Kernel A: per-row metadata
//   cat: 0 tc&&tp (mask), 1 tc&&!tp (fn1), 2 !tc&&tp (fn2), 3 !tc&&!tp (fn3)
//   is_fp <=> !tp <=> cat in {1,3};  pm = (1 - softmax(lf)[lab]) * ALP
// =====================================================================
__global__ void __launch_bounds__(256) meta_kernel(
    const float* __restrict__ lc, const float* __restrict__ lf,
    const int* __restrict__ labc, const int* __restrict__ labf,
    int* __restrict__ cat, int* __restrict__ predf, float* __restrict__ pm, int B)
{
    int b = blockIdx.x * blockDim.x + threadIdx.x;
    if (b >= B) return;
    const float* rc = lc + (size_t)b * N_COARSE;
    int pc = 0; float mc = rc[0];
    #pragma unroll
    for (int k = 1; k < N_COARSE; ++k) { float v = rc[k]; if (v > mc) { mc = v; pc = k; } }
    const float* rf = lf + (size_t)b * N_CLASS;
    int pf = 0; float mf = rf[0];
    for (int k = 1; k < N_CLASS; ++k) { float v = rf[k]; if (v > mf) { mf = v; pf = k; } }
    float se = 0.f;
    for (int k = 0; k < N_CLASS; ++k) se += __expf(rf[k] - mf);
    int lb = labf[b];
    float pf_lab = __expf(rf[lb] - mf) / se;
    bool tc = (pc == labc[b]);
    bool tp = (pf == lb);
    cat[b]   = tc ? (tp ? 0 : 1) : (tp ? 2 : 3);
    predf[b] = pf;
    pm[b]    = (1.f - pf_lab) * ALP;
}

// =====================================================================
// Kernel W: one-shot fp32 -> bf16 conversion of W (B-operand reuse:
// 8192 blocks re-read it from L2; bf16 halves that traffic and removes
// all per-iteration B-side cvt instructions).
// =====================================================================
__global__ void __launch_bounds__(256) cvtW_kernel(
    const float* __restrict__ W, __bf16* __restrict__ Wbf, int n)
{
    int i = (blockIdx.x * blockDim.x + threadIdx.x) * 4;
    if (i >= n) return;
    float4 v = *(const float4*)(W + i);
    *(v4bf*)(Wbf + i) = cvt4(v, 1.f);
}

// =====================================================================
// Kernel B: f = feature @ W^T + bias  via v_wmma_f32_16x16x32_bf16
//   grid = B/16, block = 128 (4 waves).  A tile converted to bf16 while
//   staged into LDS (8KB); each wave computes 4 column tiles; Wbf rows
//   load directly as v16bf fragments (no cvt in hot loop).
// =====================================================================
__global__ void __launch_bounds__(128) gemm_f_kernel(
    const float* __restrict__ feat, const __bf16* __restrict__ Wbf,
    const float* __restrict__ bias, float* __restrict__ f, int B)
{
    __shared__ __bf16 At[16][HDIM];                // 8 KB bf16 A tile
    const int tid   = threadIdx.x;
    const int lane  = tid & 31;
    const int wv    = tid >> 5;
    const int rbase = blockIdx.x * 16;

    // stage + convert A tile: thread t -> row t/8, cols (t%8)*32..+31
    {
        const int row = tid >> 3;
        const int c0  = (tid & 7) * 32;
        const float4* src = (const float4*)(feat + (size_t)(rbase + row) * HDIM + c0);
        #pragma unroll
        for (int j = 0; j < 8; ++j)
            *(v4bf*)&At[row][c0 + j * 4] = cvt4(src[j], 1.f);
    }
    __syncthreads();

    const int mrow  = lane & 15;
    const int ahalf = (lane & 16) ? 16 : 0;       // A-frag K start per half-wave (contiguous k = 16h + s)
    const int ncol  = lane & 15;
    const int bkb   = (lane & 16) ? 16 : 0;       // B-frag K start per half-wave (same contiguous order as A)

    v8f acc[4] = {};
    for (int k0 = 0; k0 < HDIM; k0 += 32) {
        v16bf a = frag16(*(const v8bf*)&At[mrow][k0 + ahalf],
                         *(const v8bf*)&At[mrow][k0 + ahalf + 8]);
        #pragma unroll
        for (int t = 0; t < 4; ++t) {
            const int cbase = (t * 4 + wv) * 16;
            v16bf bb = *(const v16bf*)(Wbf + (size_t)(cbase + ncol) * HDIM + k0 + bkb);
            acc[t] = __builtin_amdgcn_wmma_f32_16x16x32_bf16(false, a, false, bb,
                                                             (short)0, acc[t], false, false);
        }
    }
    #pragma unroll
    for (int t = 0; t < 4; ++t) {
        const int cbase = (t * 4 + wv) * 16;
        const float bb  = bias[cbase + ncol];
        #pragma unroll
        for (int v = 0; v < 8; ++v) {
            int m = v + ((lane & 16) ? 8 : 0);    // ISA C/D layout
            f[(size_t)(rbase + m) * HDIM + cbase + ncol] = acc[t][v] + bb;
        }
    }
}

// =====================================================================
// Kernel C: row L2 inv-norms of f + segmented class sums (scatter)
//   tabs[5][KPAD][HDIM]: 0=mask 1=fn1 2=fn2 3=fn3 4=fp ; cnts[5][KPAD]
// =====================================================================
__global__ void __launch_bounds__(256) scatter_kernel(
    const float* __restrict__ f, const int* __restrict__ cat,
    const int* __restrict__ predf, const int* __restrict__ labf,
    float* __restrict__ tabs, float* __restrict__ cnts,
    float* __restrict__ invnrm, int B)
{
    const int lane = threadIdx.x & 31;
    const int wv   = threadIdx.x >> 5;
    const int wglobal = blockIdx.x * 8 + wv;
    int r0 = wglobal * 16;
    for (int r = r0; r < r0 + 16; ++r) {
        if (r >= B) break;
        const float4* fr4 = (const float4*)(f + (size_t)r * HDIM) + lane * 2;
        float4 u = fr4[0], w = fr4[1];
        float x[8] = {u.x, u.y, u.z, u.w, w.x, w.y, w.z, w.w};
        float ss = 0.f;
        #pragma unroll
        for (int j = 0; j < 8; ++j) ss += x[j] * x[j];
        for (int o = 16; o > 0; o >>= 1) ss += __shfl_down(ss, o, 32);
        if (lane == 0) invnrm[r] = 1.f / (sqrtf(ss) + 1e-12f);
        const int c  = cat[r];
        const int lb = labf[r];
        float* t = tabs + ((size_t)c * KPAD + lb) * HDIM + lane * 8;
        #pragma unroll
        for (int j = 0; j < 8; ++j) atomicAdd(&t[j], x[j]);
        if (lane == 0) atomicAdd(&cnts[c * KPAD + lb], 1.f);
        if (c == 1 || c == 3) {                    // pred_f != lab -> fp entry
            const int p = predf[r];
            float* tfp = tabs + ((size_t)4 * KPAD + p) * HDIM + lane * 8;
            #pragma unroll
            for (int j = 0; j < 8; ++j) atomicAdd(&tfp[j], x[j]);
            if (lane == 0) atomicAdd(&cnts[4 * KPAD + p], 1.f);
        }
    }
}

// =====================================================================
// Kernel D (single block): f_mem, memn (bf16, row-normalized), g1..gp
// (row-normalized), s-flags, diversity ||cos||_F -> scal[1]
// =====================================================================
__global__ void __launch_bounds__(256) build_kernel(
    const float* __restrict__ avg_f, const float* __restrict__ tabs,
    const float* __restrict__ cnts, __bf16* __restrict__ memn_bf,
    float* __restrict__ fmem, float* __restrict__ nrm,
    float* __restrict__ gtab, float* __restrict__ sflg, float* __restrict__ scal)
{
    const int lane = threadIdx.x & 31;
    const int wv   = threadIdx.x >> 5;
    for (int k = wv; k < N_CLASS; k += 8) {
        const float cm  = cnts[0 * KPAD + k];
        const float ho  = (cm > 1e-8f) ? MOM : 1.0f;
        const float icm = 1.f / (cm + 1e-12f);
        float fm[8]; float ss = 0.f;
        #pragma unroll
        for (int j = 0; j < 8; ++j) {
            int h = lane * 8 + j;
            float fmask = tabs[((size_t)0 * KPAD + k) * HDIM + h] * icm;
            float v = avg_f[(size_t)h * N_CLASS + k] * ho + (1.f - ho) * fmask;
            fm[j] = v; ss += v * v;
        }
        for (int o = 16; o > 0; o >>= 1) ss += __shfl_down(ss, o, 32);
        ss = __shfl(ss, 0, 32);
        const float nr  = sqrtf(ss);
        const float inv = 1.f / (nr + 1e-12f);
        #pragma unroll
        for (int j = 0; j < 8; ++j) {
            int h = lane * 8 + j;
            fmem[k * HDIM + h]    = fm[j];
            memn_bf[k * HDIM + h] = (__bf16)(fm[j] * inv);
        }
        if (lane == 0) nrm[k] = nr;
        for (int t = 1; t <= 4; ++t) {
            float g[8]; float gs = 0.f;
            #pragma unroll
            for (int j = 0; j < 8; ++j) {
                int h = lane * 8 + j;
                float v = tabs[((size_t)t * KPAD + k) * HDIM + h];
                g[j] = v; gs += v * v;
            }
            for (int o = 16; o > 0; o >>= 1) gs += __shfl_down(gs, o, 32);
            gs = __shfl(gs, 0, 32);
            const float gi = 1.f / (sqrtf(gs) + 1e-12f);
            #pragma unroll
            for (int j = 0; j < 8; ++j) {
                int h = lane * 8 + j;
                gtab[((size_t)(t - 1) * KPAD + k) * HDIM + h] = g[j] * gi;
            }
            if (lane == 0) sflg[(t - 1) * KPAD + k] = (cnts[t * KPAD + k] > 1e-8f) ? 1.f : 0.f;
        }
    }
    __syncthreads();
    float acc = 0.f;
    for (int p = threadIdx.x; p < N_CLASS * N_CLASS; p += 256) {
        int k1 = p / N_CLASS, k2 = p % N_CLASS;
        const float* r1 = fmem + k1 * HDIM;
        const float* r2 = fmem + k2 * HDIM;
        float dot = 0.f;
        for (int h = 0; h < HDIM; ++h) dot += r1[h] * r2[h];
        float cs = dot / (nrm[k1] * nrm[k2]);
        acc += cs * cs;
    }
    __shared__ float red[256];
    red[threadIdx.x] = acc;
    __syncthreads();
    for (int s = 128; s > 0; s >>= 1) {
        if (threadIdx.x < s) red[threadIdx.x] += red[threadIdx.x + s];
        __syncthreads();
    }
    if (threadIdx.x == 0) scal[1] = sqrtf(red[0]);
}

// =====================================================================
// Kernel E: score GEMM (WMMA) fused with per-row deltas + 4x CE.
//   grid = B/16, block = 128.  fn_feat tile (normalization folded into
//   staging) held as bf16 in LDS; reused by WMMA phase and dot phase.
// =====================================================================
__global__ void __launch_bounds__(128) score_kernel(
    const float* __restrict__ f, const float* __restrict__ invnrm,
    const __bf16* __restrict__ memn_bf, const float* __restrict__ gtab,
    const float* __restrict__ sflg, const int* __restrict__ labf,
    const float* __restrict__ pm_arr, float* __restrict__ scal, int B)
{
    __shared__ __bf16 At[16][HDIM];                // 8 KB normalized fn tile
    __shared__ float base_s[16][KPAD];
    __shared__ float dr[16][4];
    const int tid   = threadIdx.x;
    const int lane  = tid & 31;
    const int wv    = tid >> 5;
    const int rbase = blockIdx.x * 16;

    {
        const int row = tid >> 3;
        const int c0  = (tid & 7) * 32;
        const float  inv = invnrm[rbase + row];
        const float4* src = (const float4*)(f + (size_t)(rbase + row) * HDIM + c0);
        #pragma unroll
        for (int j = 0; j < 8; ++j)
            *(v4bf*)&At[row][c0 + j * 4] = cvt4(src[j], inv);   // fn_feat
    }
    __syncthreads();

    const int mrow  = lane & 15;
    const int ahalf = (lane & 16) ? 16 : 0;
    const int ncol  = lane & 15;
    const int bkb   = (lane & 16) ? 16 : 0;
    const int cbase = wv * 16;
    const __bf16* brow = memn_bf + (size_t)(cbase + ncol) * HDIM;

    v8f acc = {};
    for (int k0 = 0; k0 < HDIM; k0 += 32) {
        v16bf a = frag16(*(const v8bf*)&At[mrow][k0 + ahalf],
                         *(const v8bf*)&At[mrow][k0 + ahalf + 8]);
        v16bf bb = *(const v16bf*)(brow + k0 + bkb);
        acc = __builtin_amdgcn_wmma_f32_16x16x32_bf16(false, a, false, bb,
                                                      (short)0, acc, false, false);
    }
    #pragma unroll
    for (int v = 0; v < 8; ++v) {
        int m = v + ((lane & 16) ? 8 : 0);
        base_s[m][cbase + ncol] = acc[v];
    }
    __syncthreads();

    // correction dots g1/g2/g3/gp[lab] . fn_feat[row]  (wave per row)
    for (int rr = 0; rr < 4; ++rr) {
        const int r  = wv * 4 + rr;
        const int b  = rbase + r;
        const int lb = labf[b];
        const int h0 = lane * 8;
        v8bf xv = *(const v8bf*)&At[r][h0];        // already normalized
        float fn[8];
        #pragma unroll
        for (int j = 0; j < 8; ++j) fn[j] = (float)xv[j];
        float d[4] = {0.f, 0.f, 0.f, 0.f};
        #pragma unroll
        for (int t = 0; t < 4; ++t) {
            const float4* g4 = (const float4*)(gtab + ((size_t)t * KPAD + lb) * HDIM + h0);
            float4 ga = g4[0], gb = g4[1];
            d[t] = ga.x * fn[0] + ga.y * fn[1] + ga.z * fn[2] + ga.w * fn[3]
                 + gb.x * fn[4] + gb.y * fn[5] + gb.z * fn[6] + gb.w * fn[7];
        }
        for (int o = 16; o > 0; o >>= 1) {
            #pragma unroll
            for (int t = 0; t < 4; ++t) d[t] += __shfl_down(d[t], o, 32);
        }
        if (lane == 0) { dr[r][0] = d[0]; dr[r][1] = d[1]; dr[r][2] = d[2]; dr[r][3] = d[3]; }
    }
    __syncthreads();

    if (tid < 16) {
        const int r  = tid;
        const int b  = rbase + r;
        const int lb = labf[b];
        const float pmv = pm_arr[b];
        float del[4];
        del[0] = ( dr[r][0] - 1.f) * pmv * sflg[0 * KPAD + lb];
        del[1] = ( dr[r][1] - 1.f) * pmv * sflg[1 * KPAD + lb];
        del[2] = ( dr[r][2] - 1.f) * pmv * sflg[2 * KPAD + lb];
        del[3] = (-dr[r][3] - 1.f) * pmv * sflg[3 * KPAD + lb];
        const float scale[4] = {1.f, 0.1f, 0.5f, 1.f};
        float loss = 0.f;
        #pragma unroll
        for (int t = 0; t < 4; ++t) {
            const float s  = scale[t] * INV_TMP;
            const float vl = (base_s[r][lb] + del[t]) * s;
            float mx = vl;
            for (int k = 0; k < N_CLASS; ++k) {
                float v = (k == lb) ? vl : base_s[r][k] * s;
                mx = fmaxf(mx, v);
            }
            float se = 0.f;
            for (int k = 0; k < N_CLASS; ++k) {
                float v = (k == lb) ? vl : base_s[r][k] * s;
                se += __expf(v - mx);
            }
            loss += mx + __logf(se) - vl;          // -log_softmax at label
        }
        atomicAdd(&scal[0], loss);
    }
}

// =====================================================================
__global__ void final_kernel(const float* __restrict__ scal, float* __restrict__ out, int B)
{
    if (threadIdx.x == 0 && blockIdx.x == 0)
        out[0] = scal[0] / (float)B + scal[1];
}

// =====================================================================
extern "C" void kernel_launch(void* const* d_in, const int* in_sizes, int n_in,
                              void* d_out, int out_size, void* d_ws, size_t ws_size,
                              hipStream_t stream)
{
    const float* feature = (const float*)d_in[0];
    const float* lc      = (const float*)d_in[1];
    const float* lf      = (const float*)d_in[2];
    const float* W       = (const float*)d_in[3];
    const float* bias    = (const float*)d_in[4];
    const float* avg_f   = (const float*)d_in[5];
    const int*   labc    = (const int*)d_in[6];
    const int*   labf    = (const int*)d_in[7];
    const int B = in_sizes[0] / HDIM;

    float* ws = (float*)d_ws;
    size_t off = 0;
    float*  f      = ws + off; off += (size_t)B * HDIM;
    float*  invnrm = ws + off; off += B;
    int*    cat    = (int*)(ws + off); off += B;
    int*    predf  = (int*)(ws + off); off += B;
    float*  pm     = ws + off; off += B;
    __bf16* Wbf    = (__bf16*)(ws + off); off += (size_t)HDIM * HDIM / 2;
    size_t zoff    = off;                       // zero-init region starts here
    float*  tabs   = ws + off; off += (size_t)5 * KPAD * HDIM;
    float*  cnts   = ws + off; off += 5 * KPAD;
    __bf16* memn   = (__bf16*)(ws + off); off += (size_t)KPAD * HDIM / 2;  // pad rows stay 0
    float*  fmem   = ws + off; off += (size_t)KPAD * HDIM;
    float*  nrm    = ws + off; off += KPAD;
    float*  gtab   = ws + off; off += (size_t)4 * KPAD * HDIM;
    float*  sflg   = ws + off; off += 4 * KPAD;
    float*  scal   = ws + off; off += 4;        // [0]=loss acc, [1]=div

    hipMemsetAsync(ws + zoff, 0, (off - zoff) * sizeof(float), stream);

    meta_kernel<<<(B + 255) / 256, 256, 0, stream>>>(lc, lf, labc, labf, cat, predf, pm, B);
    cvtW_kernel<<<(HDIM * HDIM / 4 + 255) / 256, 256, 0, stream>>>(W, Wbf, HDIM * HDIM);
    gemm_f_kernel<<<B / 16, 128, 0, stream>>>(feature, Wbf, bias, f, B);
    scatter_kernel<<<B / (16 * 8), 256, 0, stream>>>(f, cat, predf, labf, tabs, cnts, invnrm, B);
    build_kernel<<<1, 256, 0, stream>>>(avg_f, tabs, cnts, memn, fmem, nrm, gtab, sflg, scal);
    score_kernel<<<B / 16, 128, 0, stream>>>(f, invnrm, memn, gtab, sflg, labf, pm, scal, B);
    final_kernel<<<1, 1, 0, stream>>>(scal, (float*)d_out, B);
}
